// MultiHeadAttention_2138893713467
// MI455X (gfx1250) — matrix operand order repaired and checked by reading, not compiled, not run on hardware
//
#include <hip/hip_runtime.h>

// ---------- types ----------
typedef __bf16 bf16;
typedef __attribute__((ext_vector_type(4)))  __bf16 v4bf;
typedef __attribute__((ext_vector_type(8)))  __bf16 v8bf;
typedef __attribute__((ext_vector_type(16))) __bf16 v16bf;
typedef __attribute__((ext_vector_type(4)))  float  v4f;
typedef __attribute__((ext_vector_type(8)))  float  v8f;
typedef __attribute__((ext_vector_type(4)))  unsigned int v4u;
typedef __attribute__((ext_vector_type(4)))  int    v4i;
typedef __attribute__((ext_vector_type(8)))  int    v8i;

__device__ __forceinline__ v8f wmma_bf16(v16bf a, v16bf b, v8f c) {
    // D = A(16x32 bf16) x B(32x16 bf16) + C(16x16 f32)
    return __builtin_amdgcn_wmma_f32_16x16x32_bf16(
        /*neg_a=*/false, a, /*neg_b=*/false, b,
        /*c_mod=*/(short)0, c, /*reuse_a=*/false, /*reuse_b=*/false);
}

__device__ __forceinline__ v16bf cat8(v8bf lo, v8bf hi) {
    return __builtin_shufflevector(lo, hi, 0,1,2,3,4,5,6,7,8,9,10,11,12,13,14,15);
}

// ---------- Tensor Data Mover: DMA a 2-D bf16 panel (rows x 32) into LDS ----------
// Tile: tile_dim0 = 32 elements (contiguous K, 64 bytes), tile_dim1 = 64 rows,
// row stride = ldk elements.  Descriptor per CDNA5 ISA sec. 8.3-8.5.
__device__ __forceinline__ void tdm_load_panel(bf16* lds_dst, const bf16* gsrc, int ldk)
{
    unsigned long long ga = (unsigned long long)(const void*)gsrc;
    unsigned int lds = (unsigned int)(unsigned long long)lds_dst; // low 32b of flat = LDS offset

    v4u g0;
    g0[0] = 1u;                                                    // count=1 (valid user D#)
    g0[1] = lds;                                                   // lds_addr (bytes)
    g0[2] = (unsigned int)ga;                                      // global_addr[31:0]
    g0[3] = (unsigned int)((ga >> 32) & 0x1FFFFFFu) | (2u << 30);  // addr[56:32] | type=2

    v8i g1;
    g1[0] = (int)(1u << 16);          // workgroup_mask=0, data_size=1 (2 bytes)
    g1[1] = (int)(32u << 16);         // atomic_barrier_addr=0 | tensor_dim0.lo = 32
    g1[2] = (int)(64u << 16);         // tensor_dim0.hi = 0    | tensor_dim1.lo = 64
    g1[3] = (int)(32u << 16);         // tensor_dim1.hi = 0    | tile_dim0 = 32
    g1[4] = 64;                       // tile_dim1 = 64        | tile_dim2 = 0
    g1[5] = ldk;                      // tensor_dim0_stride[31:0]
    g1[6] = 0;                        // stride0.hi | tensor_dim1_stride.lo
    g1[7] = 0;                        // tensor_dim1_stride.hi

    v4i z4 = (v4i){};
#if defined(__clang_major__) && (__clang_major__ >= 23)
    v8i z8 = (v8i){};
    __builtin_amdgcn_tensor_load_to_lds(g0, g1, z4, z4, z8, 0);
#else
    __builtin_amdgcn_tensor_load_to_lds(g0, g1, z4, z4, 0);
#endif
}

// ---------- fp32 -> bf16 conversion pass (one-shot, L2-warming) ----------
__global__ __launch_bounds__(256) void f32_to_bf16_kernel(
    const float* __restrict__ x, bf16* __restrict__ y, int n)
{
    int i = (blockIdx.x * 256 + threadIdx.x) * 4;
    if (i + 3 < n) {
        v4f v = *(const v4f*)(x + i);
        v4bf o;
        o[0] = (bf16)v[0]; o[1] = (bf16)v[1]; o[2] = (bf16)v[2]; o[3] = (bf16)v[3];
        *(v4bf*)(y + i) = o;
    }
}

// ---------- GEMM: Y[m,n] = (sum_k A[m,k]*W[n,k] + bias[n]) * scale ----------
// A: bf16 [M,K] row-major.  W: bf16 [N,K] row-major (torch Linear weight).
// One workgroup = 256 rows x 64 cols (8 waves x 32 rows, shared n-range).
// Weight panel (64 rows x 32 K = 4KB) is TDM-DMA'd into double-buffered LDS.
// MODE 0: bf16 out, row-major [M,N]
// MODE 1: bf16 out, per-head transposed Vt[b][h][d][s]  (N=H*64, M=B*2048)
// MODE 2: f32  out, row-major [M,N]
template <int MODE>
__global__ __launch_bounds__(256) void gemm_bias_wmma(
    const bf16* __restrict__ A, const bf16* __restrict__ W,
    const float* __restrict__ bias, void* __restrict__ out,
    int M, int N, int K, float scale)
{
    __shared__ bf16 Wlds[2][64 * 32];          // 2 x 4KB double-buffered weight panel
    const int lane = threadIdx.x & 31;
    const int wid  = threadIdx.x >> 5;
    const int NT = N >> 6;
    const int bm = blockIdx.x / NT, bnb = blockIdx.x - bm * NT;
    const int m0 = bm * 256 + wid * 32;        // 32 rows per wave
    const int n0 = bnb << 6;                   // 64 cols per workgroup

    // A fragment addressing (16x32 bf16, lanes 0-15: K 0..15, lanes 16-31: K 16..31)
    const int am = lane & 15, ah = lane >> 4;
    const bf16* arow0 = A + (size_t)(m0 + am) * K;
    const bf16* arow1 = arow0 + (size_t)16 * K;
    // B fragment addressing (32x16 bf16: lane = col n, 16 consecutive K per lane)
    const int fn = lane & 15, fkh = lane >> 4;

    v8f acc[2][4];
#pragma unroll
    for (int mi = 0; mi < 2; ++mi)
#pragma unroll
        for (int t = 0; t < 4; ++t) acc[mi][t] = (v8f){};

    const bf16* wpanel = W + (size_t)n0 * K;   // 64 rows starting at n0
    const int nsteps = K >> 5;                 // 32-wide K steps

    if (wid == 0) tdm_load_panel(&Wlds[0][0], wpanel, K);   // prologue DMA

    for (int step = 0; step < nsteps; ++step) {
        const int cur = step & 1;
        const int k0 = step << 5;

        __syncthreads();  // buffer (cur^1) no longer being read -> safe to overwrite
        if (wid == 0) {
            if (step + 1 < nsteps) {
                tdm_load_panel(&Wlds[cur ^ 1][0], wpanel + (k0 + 32), K);
                __builtin_amdgcn_s_wait_tensorcnt(1);   // current panel complete
            } else {
                __builtin_amdgcn_s_wait_tensorcnt(0);
            }
        }
        __syncthreads();  // current panel visible to all waves

        v16bf a0 = cat8(*(const v8bf*)(arow0 + k0 + ah * 16),
                        *(const v8bf*)(arow0 + k0 + 8 + ah * 16));
        v16bf a1 = cat8(*(const v8bf*)(arow1 + k0 + ah * 16),
                        *(const v8bf*)(arow1 + k0 + 8 + ah * 16));
        const bf16* wp = &Wlds[cur][0];
#pragma unroll
        for (int t = 0; t < 4; ++t) {
            v16bf b = *(const v16bf*)(wp + (t * 16 + fn) * 32 + fkh * 16);
            acc[0][t] = wmma_bf16(a0, b, acc[0][t]);
            acc[1][t] = wmma_bf16(a1, b, acc[1][t]);
        }
    }

    // Epilogue. C layout: lane holds col n = lane&15, rows (lane>>4)*8 + r.
    const int cn = lane & 15, chi = lane >> 4;
#pragma unroll
    for (int t = 0; t < 4; ++t) {
        const int n = n0 + t * 16 + cn;
        const float bv = bias[n];
#pragma unroll
        for (int mi = 0; mi < 2; ++mi) {
            const int mbase = m0 + mi * 16 + chi * 8;
            if (MODE == 2) {
                float* o = (float*)out;
#pragma unroll
                for (int r = 0; r < 8; ++r)
                    o[(size_t)(mbase + r) * N + n] = (acc[mi][t][r] + bv) * scale;
            } else if (MODE == 0) {
                bf16* o = (bf16*)out;
#pragma unroll
                for (int r = 0; r < 8; ++r)
                    o[(size_t)(mbase + r) * N + n] = (bf16)((acc[mi][t][r] + bv) * scale);
            } else { // MODE 1: Vt[b][h][d][s], contiguous over the 8 rows (s)
                const int h = n >> 6, d = n & 63;
                const int b = mbase >> 11, s = mbase & 2047;
                v8bf st;
#pragma unroll
                for (int r = 0; r < 8; ++r) st[r] = (bf16)((acc[mi][t][r] + bv) * scale);
                bf16* o = (bf16*)out + (((size_t)(b * 16 + h) * 64 + d) << 11) + s;
                *(v8bf*)o = st;
            }
        }
    }
}

// ---------- flash attention: one wave = 16 queries of one (b,h), streams keys ----------
// Qp,Kp: bf16 [B,2048,1024] (head slice h*64..h*64+63); Qp pre-scaled by 1/sqrt(64).
// Vt: bf16 [B][16][64][2048].  Ctx out: bf16 [B,2048,1024].
__global__ __launch_bounds__(256) void mha_flash_wmma(
    const bf16* __restrict__ Qp, const bf16* __restrict__ Kp,
    const bf16* __restrict__ Vt, bf16* __restrict__ Ctx)
{
    __shared__ bf16 Plds[8 * 16 * 32];      // 1KB per wave, 8 waves
    const int lane = threadIdx.x & 31;
    const int wid  = threadIdx.x >> 5;
    const int wave = blockIdx.x * 8 + wid;  // 4096 waves total
    const int qt = wave & 127;              // 128 q-tiles of 16 rows
    const int bh = wave >> 7;               // 0..31
    const int b = bh >> 4, h = bh & 15;

    const int am = lane & 15, ah = lane >> 4;  // A-frag / C-row mapping
    const int bn = lane & 15, bkh = lane >> 4; // B-frag mapping

    // load the two Q A-fragments (d = 0..31, 32..63) once
    const bf16* qrow = Qp + (size_t)(b * 2048 + qt * 16 + am) * 1024 + h * 64;
    v16bf qf[2];
#pragma unroll
    for (int db = 0; db < 2; ++db)
        qf[db] = cat8(*(const v8bf*)(qrow + db * 32 + ah * 16),
                      *(const v8bf*)(qrow + db * 32 + 8 + ah * 16));

    const bf16* kbase = Kp + (size_t)b * 2048 * 1024 + h * 64;
    const bf16* vbase = Vt + (size_t)bh * 64 * 2048;
    bf16* pl = Plds + wid * 16 * 32;

    float mrow[8], lrow[8];
    v8f o[4];
#pragma unroll
    for (int r = 0; r < 8; ++r) { mrow[r] = -3.0e38f; lrow[r] = 0.0f; }
#pragma unroll
    for (int dt = 0; dt < 4; ++dt) o[dt] = (v8f){};

    for (int k0 = 0; k0 < 2048; k0 += 32) {
        // ---- scores: two 16x16 tiles over 32 keys ----
        v8f s[2];
#pragma unroll
        for (int nt = 0; nt < 2; ++nt) {
            const bf16* krow = kbase + (size_t)(k0 + nt * 16 + bn) * 1024;
            v16bf kf0 = *(const v16bf*)(krow + bkh * 16);
            v16bf kf1 = *(const v16bf*)(krow + 32 + bkh * 16);
            v8f z = (v8f){};
            z = wmma_bf16(qf[0], kf0, z);
            z = wmma_bf16(qf[1], kf1, z);
            s[nt] = z;
        }
        // ---- online softmax (rows live in 8 regs/lane; cols across 16 lanes) ----
        float mc[8];
#pragma unroll
        for (int r = 0; r < 8; ++r) mc[r] = fmaxf(s[0][r], s[1][r]);
#pragma unroll
        for (int d = 1; d <= 8; d <<= 1)
#pragma unroll
            for (int r = 0; r < 8; ++r) mc[r] = fmaxf(mc[r], __shfl_xor(mc[r], d, 32));
        float alpha[8], rs[8];
#pragma unroll
        for (int r = 0; r < 8; ++r) {
            float mn = fmaxf(mrow[r], mc[r]);
            alpha[r] = __expf(mrow[r] - mn);
            mrow[r] = mn;
            rs[r] = 0.0f;
        }
#pragma unroll
        for (int nt = 0; nt < 2; ++nt)
#pragma unroll
            for (int r = 0; r < 8; ++r) {
                float p = __expf(s[nt][r] - mrow[r]);
                rs[r] += p;
                pl[(ah * 8 + r) * 32 + nt * 16 + am] = (bf16)p;  // C-layout -> row-major tile
            }
#pragma unroll
        for (int d = 1; d <= 8; d <<= 1)
#pragma unroll
            for (int r = 0; r < 8; ++r) rs[r] += __shfl_xor(rs[r], d, 32);
#pragma unroll
        for (int r = 0; r < 8; ++r) lrow[r] = lrow[r] * alpha[r] + rs[r];
#pragma unroll
        for (int dt = 0; dt < 4; ++dt)
#pragma unroll
            for (int r = 0; r < 8; ++r) o[dt][r] *= alpha[r];

        // LDS stores -> loads (same wave, different lanes): order + compiler fence
        asm volatile("s_wait_dscnt 0x0" ::: "memory");
        v16bf pf = cat8(*(const v8bf*)(pl + am * 32 + ah * 16),
                        *(const v8bf*)(pl + am * 32 + 8 + ah * 16));
        asm volatile("" ::: "memory");

        // ---- PV: accumulate 16x64 output ----
#pragma unroll
        for (int dt = 0; dt < 4; ++dt) {
            v16bf vf = *(const v16bf*)(vbase + (size_t)(dt * 16 + bn) * 2048 + k0 + bkh * 16);
            o[dt] = wmma_bf16(pf, vf, o[dt]);
        }
    }

    // ---- normalize and store context [b, s, h*64+d] ----
    float inv[8];
#pragma unroll
    for (int r = 0; r < 8; ++r) inv[r] = 1.0f / lrow[r];
    bf16* crow = Ctx + (size_t)(b * 2048 + qt * 16) * 1024 + h * 64;
#pragma unroll
    for (int dt = 0; dt < 4; ++dt)
#pragma unroll
        for (int r = 0; r < 8; ++r)
            crow[(size_t)(ah * 8 + r) * 1024 + dt * 16 + am] = (bf16)(o[dt][r] * inv[r]);
}

// ---------- host launch ----------
extern "C" void kernel_launch(void* const* d_in, const int* in_sizes, int n_in,
                              void* d_out, int out_size, void* d_ws, size_t ws_size,
                              hipStream_t stream) {
    const float* q   = (const float*)d_in[0];
    const float* k   = (const float*)d_in[1];
    const float* v   = (const float*)d_in[2];
    const float* w_q = (const float*)d_in[3];
    const float* b_q = (const float*)d_in[4];
    const float* w_k = (const float*)d_in[5];
    const float* b_k = (const float*)d_in[6];
    const float* w_v = (const float*)d_in[7];
    const float* b_v = (const float*)d_in[8];
    const float* w_o = (const float*)d_in[9];
    const float* b_o = (const float*)d_in[10];

    const int M = 4096, D = 1024;
    const int N_ACT = M * D;       // 4,194,304
    const int N_W   = D * D;       // 1,048,576
    const size_t MB = (size_t)1 << 20;

    char* ws = (char*)d_ws;
    bf16* qb  = (bf16*)(ws + 0 * MB);
    bf16* kb  = (bf16*)(ws + 8 * MB);
    bf16* vb  = (bf16*)(ws + 16 * MB);
    bf16* wqb = (bf16*)(ws + 24 * MB);
    bf16* wkb = (bf16*)(ws + 26 * MB);
    bf16* wvb = (bf16*)(ws + 28 * MB);
    bf16* wob = (bf16*)(ws + 30 * MB);
    bf16* Qp  = (bf16*)(ws + 32 * MB);
    bf16* Kp  = (bf16*)(ws + 40 * MB);
    bf16* Vt  = (bf16*)(ws + 48 * MB);
    bf16* Ctx = (bf16*)(ws + 56 * MB);

    // fp32 -> bf16
    f32_to_bf16_kernel<<<N_ACT / 1024, 256, 0, stream>>>(q, qb, N_ACT);
    f32_to_bf16_kernel<<<N_ACT / 1024, 256, 0, stream>>>(k, kb, N_ACT);
    f32_to_bf16_kernel<<<N_ACT / 1024, 256, 0, stream>>>(v, vb, N_ACT);
    f32_to_bf16_kernel<<<N_W / 1024, 256, 0, stream>>>(w_q, wqb, N_W);
    f32_to_bf16_kernel<<<N_W / 1024, 256, 0, stream>>>(w_k, wkb, N_W);
    f32_to_bf16_kernel<<<N_W / 1024, 256, 0, stream>>>(w_v, wvb, N_W);
    f32_to_bf16_kernel<<<N_W / 1024, 256, 0, stream>>>(w_o, wob, N_W);

    // projections: (M/256)*(N/64) = 16*16 = 256 workgroups of 8 waves
    gemm_bias_wmma<0><<<256, 256, 0, stream>>>(qb, wqb, b_q, (void*)Qp, M, D, D, 0.125f); // 1/sqrt(64)
    gemm_bias_wmma<0><<<256, 256, 0, stream>>>(kb, wkb, b_k, (void*)Kp, M, D, D, 1.0f);
    gemm_bias_wmma<1><<<256, 256, 0, stream>>>(vb, wvb, b_v, (void*)Vt, M, D, D, 1.0f);

    // attention: 32 (b,h) * 128 q-tiles = 4096 waves -> 512 blocks
    mha_flash_wmma<<<512, 256, 0, stream>>>(Qp, Kp, Vt, Ctx);

    // output projection -> fp32 d_out
    gemm_bias_wmma<2><<<256, 256, 0, stream>>>(Ctx, wob, b_o, d_out, M, D, D, 1.0f);
}
